// Encoder_33964601377143
// MI455X (gfx1250) — matrix operand order repaired and checked by reading, not compiled, not run on hardware
//
#include <hip/hip_runtime.h>
#include <cstdint>

typedef __attribute__((ext_vector_type(16))) __bf16        v16bf;
typedef __attribute__((ext_vector_type(8)))  float         v8f;
typedef __attribute__((ext_vector_type(8)))  unsigned int  v8u;

// ---------- bf16 <-> f32 helpers (bits only; RNE) ----------
__device__ __forceinline__ unsigned short f2bf(float f) {
  union { float f; unsigned int u; } a; a.f = f;
  unsigned int r = a.u + 0x7FFFu + ((a.u >> 16) & 1u);
  return (unsigned short)(r >> 16);
}
__device__ __forceinline__ float bf2f(unsigned short h) {
  union { unsigned int u; float f; } a; a.u = ((unsigned int)h) << 16;
  return a.f;
}
__device__ __forceinline__ float ldA(const float* p)          { return *p; }
__device__ __forceinline__ float ldA(const unsigned short* p) { return bf2f(*p); }
__device__ __forceinline__ void stC(float* p, float v)        { *p = v; }
__device__ __forceinline__ void stC(unsigned short* p, float v){ *p = f2bf(v); }

__device__ __forceinline__ v8f vzero() {
  v8f z;
#pragma unroll
  for (int i = 0; i < 8; i++) z[i] = 0.f;
  return z;
}

__device__ __forceinline__ v8f wmma_bf16(v8u a, v8u b, v8f c) {
  return __builtin_amdgcn_wmma_f32_16x16x32_bf16(
      false, __builtin_bit_cast(v16bf, a),
      false, __builtin_bit_cast(v16bf, b),
      (short)0, c, false, false);
}

// =====================================================================
// GEMM: C[M,N] = act(A[M,K] @ W[K,N] + bias), bf16 WMMA, f32 accumulate.
// Block tile 128x128, 8 waves (4Mx2N), wave tile 32x64, K-step 32.
// 2-stage software pipeline: double-buffered LDS, next tile staged in
// registers while current tile's WMMAs execute.
// =====================================================================
template <typename AT, typename OT, bool RELU>
__global__ __launch_bounds__(256)
void gemm_wmma_kernel(const AT* __restrict__ A, const float* __restrict__ W,
                      const float* __restrict__ bias, OT* __restrict__ Cout,
                      int M, int N, int K) {
  __shared__ unsigned short As[2][128][32];   // [buf][m][k] bf16
  __shared__ unsigned short Bs[2][128][32];   // [buf][n][k] bf16 (transposed)

  const int bm = blockIdx.y * 128;
  const int bn = blockIdx.x * 128;
  const int tid = threadIdx.x;
  const int w = tid >> 5, l = tid & 31;
  const int lm = l & 15, lh = l >> 4;
  const int wm = (w >> 1) * 32;               // wave M offset: 0,32,64,96
  const int wn = (w & 1) * 64;                // wave N offset: 0,64

  // staging coordinates (constant per thread)
  const int ar = tid >> 3, ac = (tid & 7) * 4;       // A rows, 32 apart per t
  const int bk = tid >> 5, bn4 = (tid & 31) * 4;     // B k-rows, 8 apart per t

  v8f acc[2][4];
#pragma unroll
  for (int i = 0; i < 2; i++)
#pragma unroll
    for (int j = 0; j < 4; j++) acc[i][j] = vzero();

  float aReg[16], bReg[16];
  const int KT = K >> 5;

  auto loadG = [&](int kt) {
    const int k0 = kt * 32;
#pragma unroll
    for (int t = 0; t < 4; t++) {
      const AT* p = A + (size_t)(bm + ar + t * 32) * K + (k0 + ac);
#pragma unroll
      for (int j = 0; j < 4; j++) aReg[t * 4 + j] = ldA(p + j);
    }
#pragma unroll
    for (int t = 0; t < 4; t++) {
      const float* p = W + (size_t)(k0 + bk + t * 8) * N + (bn + bn4);
#pragma unroll
      for (int j = 0; j < 4; j++) bReg[t * 4 + j] = p[j];
    }
  };
  auto storeL = [&](int buf) {
#pragma unroll
    for (int t = 0; t < 4; t++)
#pragma unroll
      for (int j = 0; j < 4; j++) As[buf][ar + t * 32][ac + j] = f2bf(aReg[t * 4 + j]);
#pragma unroll
    for (int t = 0; t < 4; t++)
#pragma unroll
      for (int j = 0; j < 4; j++) Bs[buf][bn4 + j][bk + t * 8] = f2bf(bReg[t * 4 + j]);
  };
  auto compute = [&](int buf) {
    v8u af[2], bfr[4];
#pragma unroll
    for (int i = 0; i < 2; i++) {
      const unsigned int* a32 = (const unsigned int*)(&As[buf][wm + i * 16 + lm][0]);
#pragma unroll
      for (int j = 0; j < 4; j++) {
        af[i][j]     = a32[4 * lh + j];       // K = 8h+2j, +1
        af[i][4 + j] = a32[8 + 4 * lh + j];   // K = 16+8h+2j, +1
      }
    }
#pragma unroll
    for (int i = 0; i < 4; i++) {
      const unsigned int* b32 = (const unsigned int*)(&Bs[buf][wn + i * 16 + lm][0]);
#pragma unroll
      for (int j = 0; j < 8; j++) bfr[i][j] = b32[(j >> 2) * 8 + 4 * lh + (j & 3)];
    }
#pragma unroll
    for (int i = 0; i < 2; i++)
#pragma unroll
      for (int j = 0; j < 4; j++)
        acc[i][j] = wmma_bf16(af[i], bfr[j], acc[i][j]);
  };

  // prologue
  loadG(0);
  storeL(0);
  __syncthreads();

  for (int kt = 0; kt < KT; kt++) {
    const int cur = kt & 1;
    if (kt + 1 < KT) loadG(kt + 1);   // global fetch overlaps WMMAs below
    compute(cur);
    if (kt + 1 < KT) {
      __syncthreads();                // all reads of buffer 1-cur are done
      storeL(1 - cur);
      __syncthreads();                // stores visible before next compute
    }
  }

  // epilogue: C/D layout — n = lane&15, M = 8*(lane>>4)+r
#pragma unroll
  for (int i = 0; i < 2; i++) {
#pragma unroll
    for (int j = 0; j < 4; j++) {
      int gn = bn + wn + j * 16 + lm;
      float bv = bias[gn];
#pragma unroll
      for (int r = 0; r < 8; r++) {
        int gm = bm + wm + i * 16 + lh * 8 + r;
        float vv = acc[i][j][r] + bv;
        if (RELU) vv = fmaxf(vv, 0.f);
        stC(&Cout[(size_t)gm * N + gn], vv);
      }
    }
  }
}

// =====================================================================
// Window attention: one block per (window, head). N=144 tokens, HD=64.
// K/V + per-head bias table resident in LDS, 9 query chunks of 16 rows,
// WMMA for QK^T and PV. Softmax 16 rows x 16 col-groups. KV pad 144->160.
// =====================================================================
__global__ __launch_bounds__(256)
void win_attn_kernel(const unsigned short* __restrict__ q,
                     const unsigned short* __restrict__ k,
                     const unsigned short* __restrict__ v,
                     const float* __restrict__ rpb,
                     float* __restrict__ out) {
  __shared__ unsigned short ks_[144][64];   // K row-major [n][d]
  __shared__ unsigned short vsT[64][160];   // V transposed [d][n], n padded->160
  __shared__ unsigned short qs_[16][64];    // Q chunk
  __shared__ float          ss[16][160];    // scores f32
  __shared__ unsigned short ps[16][160];    // probabilities bf16
  __shared__ float          red2[16][17];   // softmax partials (padded)
  __shared__ float          rpbs[529];      // rel-pos bias, this head's slice

  const int blk = blockIdx.x;
  const int bw = blk >> 3;                  // batch*window (0..255)
  const int hh = blk & 7;                   // head
  const int b = bw >> 2;
  const int wwin = bw & 3;
  const int tid = threadIdx.x;
  const int w = tid >> 5, l = tid & 31, lm = l & 15, lh = l >> 4;
  const int srow = tid & 15, scg = tid >> 4;  // softmax row / col-group

  auto rowmap = [&](int n) {
    int iy = n / 12, ix = n - iy * 12;
    return b * 576 + ((wwin >> 1) * 12 + iy) * 24 + (wwin & 1) * 12 + ix;
  };

  for (int idx = tid; idx < 144 * 64; idx += 256) {
    int n = idx >> 6, c = idx & 63;
    size_t off = (size_t)rowmap(n) * 512 + hh * 64 + c;
    ks_[n][c] = k[off];
    vsT[c][n] = v[off];
  }
  for (int idx = tid; idx < 64 * 16; idx += 256)
    vsT[idx >> 4][144 + (idx & 15)] = 0;   // zero pad cols 144..159
  for (int idx = tid; idx < 529; idx += 256)
    rpbs[idx] = rpb[idx * 8 + hh];         // stage bias table into LDS
  __syncthreads();

  const float scale = 0.125f;              // 1/sqrt(64)

  for (int qc = 0; qc < 9; qc++) {
    for (int idx = tid; idx < 16 * 64; idx += 256) {
      int n = idx >> 6, c = idx & 63;
      qs_[n][c] = q[(size_t)rowmap(qc * 16 + n) * 512 + hh * 64 + c];
    }
    __syncthreads();

    // query-row window coords for this chunk (invariant over col tiles)
    int qiy[8], qix[8];
#pragma unroll
    for (int r = 0; r < 8; r++) {
      int qi = qc * 16 + lh * 8 + r;
      qiy[r] = qi / 12;
      qix[r] = qi - qiy[r] * 12;
    }

    // S chunk = Q(16x64) @ K^T(64x144), col tiles across waves
    for (int ct = w; ct < 9; ct += 8) {
      v8f acc = vzero();
#pragma unroll
      for (int kk = 0; kk < 2; kk++) {
        v8u af, bfr;
        const unsigned int* a32 = (const unsigned int*)(&qs_[lm][kk * 32]);
#pragma unroll
        for (int j = 0; j < 4; j++) { af[j] = a32[4 * lh + j]; af[4 + j] = a32[8 + 4 * lh + j]; }
        const unsigned int* b32 = (const unsigned int*)(&ks_[ct * 16 + lm][kk * 32]);
#pragma unroll
        for (int j = 0; j < 8; j++) bfr[j] = b32[(j >> 2) * 8 + 4 * lh + (j & 3)];
        acc = wmma_bf16(af, bfr, acc);
      }
      int jc = ct * 16 + lm;
      int jy = jc / 12, jx = jc - jy * 12;
#pragma unroll
      for (int r = 0; r < 8; r++) {
        int dy = qiy[r] - jy + 11;
        int dx = qix[r] - jx + 11;
        ss[lh * 8 + r][jc] = acc[r] * scale + rpbs[dy * 23 + dx];
      }
    }
    __syncthreads();

    // ---- parallel softmax: 16 rows x 16 col-groups ----
    float mx = -1e30f;
    for (int j = scg; j < 144; j += 16) mx = fmaxf(mx, ss[srow][j]);
    red2[srow][scg] = mx;
    __syncthreads();
#pragma unroll
    for (int i = 0; i < 16; i++) mx = fmaxf(mx, red2[srow][i]);
    float sum = 0.f;
    for (int j = scg; j < 144; j += 16) {
      float e = __expf(ss[srow][j] - mx);
      ss[srow][j] = e;
      sum += e;
    }
    __syncthreads();
    red2[srow][scg] = sum;
    __syncthreads();
    sum = 0.f;
#pragma unroll
    for (int i = 0; i < 16; i++) sum += red2[srow][i];
    float inv = 1.f / sum;
    for (int j = scg; j < 160; j += 16)
      ps[srow][j] = (j < 144) ? f2bf(ss[srow][j] * inv) : (unsigned short)0;
    __syncthreads();

    // O chunk = P(16x160) @ V(160x64), 4 col tiles on waves 0..3
    if (w < 4) {
      v8f acc = vzero();
#pragma unroll
      for (int kk = 0; kk < 5; kk++) {
        v8u af, bfr;
        const unsigned int* a32 = (const unsigned int*)(&ps[lm][0]);
#pragma unroll
        for (int j = 0; j < 4; j++) {
          af[j]     = a32[kk * 16 + 4 * lh + j];
          af[4 + j] = a32[kk * 16 + 8 + 4 * lh + j];
        }
        const unsigned int* b32 = (const unsigned int*)(&vsT[w * 16 + lm][0]);
#pragma unroll
        for (int j = 0; j < 8; j++) bfr[j] = b32[kk * 16 + (j >> 2) * 8 + 4 * lh + (j & 3)];
        acc = wmma_bf16(af, bfr, acc);
      }
#pragma unroll
      for (int r = 0; r < 8; r++) {
        int orow = qc * 16 + lh * 8 + r;
        int oc = hh * 64 + w * 16 + lm;
        out[(size_t)rowmap(orow) * 512 + oc] = acc[r];  // window-reverse folded in
      }
    }
    __syncthreads();
  }
}

// =====================================================================
// Fused residual + LayerNorm over C=512, one block per row.
// =====================================================================
__global__ __launch_bounds__(256)
void add_ln_kernel(const float* __restrict__ a, const float* __restrict__ res,
                   const float* __restrict__ g, const float* __restrict__ be,
                   float* __restrict__ out) {
  __shared__ float red[256];
  const int row = blockIdx.x;
  const int tid = threadIdx.x;
  const size_t base = (size_t)row * 512;
  float v0 = a[base + tid] + res[base + tid];
  float v1 = a[base + tid + 256] + res[base + tid + 256];
  red[tid] = v0 + v1;
  __syncthreads();
  for (int off = 128; off > 0; off >>= 1) { if (tid < off) red[tid] += red[tid + off]; __syncthreads(); }
  float mean = red[0] * (1.f / 512.f);
  __syncthreads();
  float d0 = v0 - mean, d1 = v1 - mean;
  red[tid] = d0 * d0 + d1 * d1;
  __syncthreads();
  for (int off = 128; off > 0; off >>= 1) { if (tid < off) red[tid] += red[tid + off]; __syncthreads(); }
  float rstd = rsqrtf(red[0] * (1.f / 512.f) + 1e-5f);
  out[base + tid]       = d0 * rstd * g[tid]       + be[tid];
  out[base + tid + 256] = d1 * rstd * g[tid + 256] + be[tid + 256];
}

// =====================================================================
extern "C" void kernel_launch(void* const* d_in, const int* in_sizes, int n_in,
                              void* d_out, int out_size, void* d_ws, size_t ws_size,
                              hipStream_t stream) {
  (void)in_sizes; (void)n_in; (void)out_size; (void)ws_size;
  const int M = 36864, Cd = 512, FF = 2048;

  const float* x   = (const float*)d_in[0];
  const float* Wq  = (const float*)d_in[1];
  const float* bq  = (const float*)d_in[2];
  const float* Wk  = (const float*)d_in[3];
  const float* bk  = (const float*)d_in[4];
  const float* Wv  = (const float*)d_in[5];
  const float* bv  = (const float*)d_in[6];
  const float* Wo  = (const float*)d_in[7];
  const float* bo  = (const float*)d_in[8];
  const float* rpb = (const float*)d_in[9];
  const float* g1  = (const float*)d_in[10];
  const float* b1  = (const float*)d_in[11];
  const float* W1  = (const float*)d_in[12];
  const float* bf1 = (const float*)d_in[13];
  const float* W2  = (const float*)d_in[14];
  const float* bf2 = (const float*)d_in[15];
  const float* g2  = (const float*)d_in[16];
  const float* b2  = (const float*)d_in[17];

  char* wsb = (char*)d_ws;
  size_t off = 0;
  auto alloc = [&](size_t bytes) -> char* {
    char* p = wsb + off; off += (bytes + 255) & ~(size_t)255; return p;
  };
  float* xb            = (float*)alloc((size_t)M * Cd * 4);
  float* t0            = (float*)alloc((size_t)M * Cd * 4);
  float* attn          = (float*)alloc((size_t)M * Cd * 4);
  unsigned short* qb   = (unsigned short*)alloc((size_t)M * Cd * 2);
  unsigned short* kb   = (unsigned short*)alloc((size_t)M * Cd * 2);
  unsigned short* vb   = (unsigned short*)alloc((size_t)M * Cd * 2);
  unsigned short* hb   = (unsigned short*)alloc((size_t)M * FF * 2);

  hipMemcpyAsync(xb, x, (size_t)M * Cd * 4, hipMemcpyDeviceToDevice, stream);

  dim3 blk(256);
  dim3 gC(Cd / 128, M / 128);   // (4, 288)
  dim3 gF(FF / 128, M / 128);   // (16, 288)

  for (int l = 0; l < 3; l++) {
    const float* Wq_l = Wq + (size_t)l * Cd * Cd;
    const float* Wk_l = Wk + (size_t)l * Cd * Cd;
    const float* Wv_l = Wv + (size_t)l * Cd * Cd;
    const float* Wo_l = Wo + (size_t)l * Cd * Cd;
    const float* bq_l = bq + (size_t)l * Cd;
    const float* bk_l = bk + (size_t)l * Cd;
    const float* bv_l = bv + (size_t)l * Cd;
    const float* bo_l = bo + (size_t)l * Cd;
    const float* rpb_l = rpb + (size_t)l * 529 * 8;
    const float* g1_l = g1 + (size_t)l * Cd;
    const float* b1_l = b1 + (size_t)l * Cd;
    const float* W1_l = W1 + (size_t)l * Cd * FF;
    const float* bf1_l = bf1 + (size_t)l * FF;
    const float* W2_l = W2 + (size_t)l * FF * Cd;
    const float* bf2_l = bf2 + (size_t)l * Cd;
    const float* g2_l = g2 + (size_t)l * Cd;
    const float* b2_l = b2 + (size_t)l * Cd;

    gemm_wmma_kernel<float, unsigned short, false><<<gC, blk, 0, stream>>>(xb, Wq_l, bq_l, qb, M, Cd, Cd);
    gemm_wmma_kernel<float, unsigned short, false><<<gC, blk, 0, stream>>>(xb, Wk_l, bk_l, kb, M, Cd, Cd);
    gemm_wmma_kernel<float, unsigned short, false><<<gC, blk, 0, stream>>>(xb, Wv_l, bv_l, vb, M, Cd, Cd);
    win_attn_kernel<<<dim3(2048), blk, 0, stream>>>(qb, kb, vb, rpb_l, attn);
    gemm_wmma_kernel<float, float, false><<<gC, blk, 0, stream>>>(attn, Wo_l, bo_l, t0, M, Cd, Cd);
    add_ln_kernel<<<dim3(M), blk, 0, stream>>>(t0, xb, g1_l, b1_l, xb);
    gemm_wmma_kernel<float, unsigned short, true><<<gF, blk, 0, stream>>>(xb, W1_l, bf1_l, hb, M, FF, Cd);
    gemm_wmma_kernel<unsigned short, float, false><<<gC, blk, 0, stream>>>(hb, W2_l, bf2_l, t0, M, Cd, FF);
    float* dst = (l == 2) ? (float*)d_out : xb;
    add_ln_kernel<<<dim3(M), blk, 0, stream>>>(t0, xb, g2_l, b2_l, dst);
  }
}
